// PointerDecoder_12756052869475
// MI455X (gfx1250) — compile-verified, matrix operand order checked by reading, not run on hardware
//
#include <hip/hip_runtime.h>
#include <hip/hip_bf16.h>
#include <math.h>

#define B_ 512
#define N_ 256
#define D_ 256
#define NEG_INF_ (-1e30f)

typedef __attribute__((ext_vector_type(16))) __bf16 v16bf;
typedef __attribute__((ext_vector_type(8)))  float  v8f;
typedef unsigned short u16;
typedef unsigned int   u32;

// ---------------- helpers ----------------
__device__ __forceinline__ u16 f2bf(float x){
  u32 u = __float_as_uint(x);
  u32 r = (u + 0x7FFFu + ((u >> 16) & 1u)) >> 16;   // round-to-nearest-even
  return (u16)r;
}

// Load one 16x32 bf16 WMMA fragment (A-layout / B-layout symmetric form).
// p points at (row_or_col)*stride + kbase; hlf selects the lane-half K subset.
__device__ __forceinline__ v16bf load_frag(const u16* __restrict__ p, int hlf){
  union { v16bf v; u32 u[8]; } f;
  const u16* q = p + 8*hlf;
#pragma unroll
  for (int j = 0; j < 8; ++j){
    int off = 2*j + ((j >= 4) ? 8 : 0);     // K pairs: 0..7 then 16..23 (rel.)
    f.u[j] = *(const u32*)(q + off);
  }
  return f.v;
}

__device__ __forceinline__ v8f wmma_bf16(v16bf a, v16bf b, v8f c){
  return __builtin_amdgcn_wmma_f32_16x16x32_bf16(
      false, a, false, b, (short)0, c, false, false);
}

__device__ __forceinline__ float sigmoidf_(float x){ return 1.0f / (1.0f + expf(-x)); }

// ---------------- f32 -> bf16 conversion ----------------
__global__ void k_to_bf16(const float* __restrict__ src, u16* __restrict__ dst, int n){
  for (int i = blockIdx.x*blockDim.x + threadIdx.x; i < n; i += gridDim.x*blockDim.x)
    dst[i] = f2bf(src[i]);
}

// ---------------- keys = enc @ wk^T + bk  (bf16 WMMA GEMM) ----------------
__global__ __launch_bounds__(256) void k_keys_gemm(
    const u16* __restrict__ encb, const u16* __restrict__ wkb,
    const float* __restrict__ bk, u16* __restrict__ keysb){
  const int wave = threadIdx.x >> 5;
  const int lane = threadIdx.x & 31;
  const int hlf  = lane >> 4;
  const int lrow = lane & 15;
  const size_t m0 = (size_t)blockIdx.x*128 + (size_t)wave*16;   // 8 waves x 16 rows

  v16bf a[8];
#pragma unroll
  for (int kc = 0; kc < 8; ++kc)
    a[kc] = load_frag(encb + (m0 + lrow)*D_ + kc*32, hlf);

#pragma unroll 1
  for (int t = 0; t < 16; ++t){
    v8f acc = {0.f,0.f,0.f,0.f,0.f,0.f,0.f,0.f};
#pragma unroll
    for (int kc = 0; kc < 8; ++kc)
      acc = wmma_bf16(a[kc], load_frag(wkb + (t*16 + lrow)*D_ + kc*32, hlf), acc);
    float bkv = bk[t*16 + lrow];
#pragma unroll
    for (int r = 0; r < 8; ++r){
      int m = r + 8*hlf;
      keysb[(m0 + m)*D_ + t*16 + lrow] = f2bf(acc[r] + bkv);
    }
  }
}

// ---------------- persistent pointer-decoder: 16 batch rows / WG ----------------
__global__ __launch_bounds__(512) void k_decode(
    const float* __restrict__ enc, const float* __restrict__ init_tok,
    const float* __restrict__ b_ih, const float* __restrict__ b_hh,
    const float* __restrict__ bq,
    const u16* __restrict__ encb, const u16* __restrict__ keysb,
    const u16* __restrict__ wihb, const u16* __restrict__ whhb,
    const u16* __restrict__ wqb,
    float* __restrict__ out)
{
  extern __shared__ unsigned char smem[];
  u16*   Xb = (u16*)smem;                       // [16][256] prev_embed bf16
  u16*   Hb = Xb + 16*256;                      // [16][256] h bf16
  u16*   Qb = Hb + 16*256;                      // [16][256] query bf16
  float* Hf = (float*)(Qb + 16*256);            // [16][256] h f32
  float* GI = Hf + 16*256;                      // [16][768] x @ w_ih^T
  float* GH = GI + 16*768;                      // [16][768] h @ w_hh^T
  float* LG = GH + 16*768;                      // [16][256] logits
  unsigned char* MK = (unsigned char*)(LG + 16*256); // [16][256] mask

  const int tid  = threadIdx.x;
  const int wave = tid >> 5;
  const int lane = tid & 31;
  const int hlf  = lane >> 4;
  const int lrow = lane & 15;
  const int bg0  = blockIdx.x * 16;

  // ---- prologue: mask=1, X0=init_token, h0 = mean(enc over n) ----
  for (int v = tid; v < 16*256; v += 512){
    int d = v & 255;
    MK[v] = 1;
    Xb[v] = f2bf(init_tok[d]);
  }
#pragma unroll 1
  for (int j = 0; j < 8; ++j){
    int v = tid + 512*j;
    int b = v >> 8, d = v & 255;
    const float* p = enc + ((size_t)(bg0 + b))*(size_t)N_*D_ + d;
    float s = 0.f;
#pragma unroll 1
    for (int n = 0; n < N_; ++n) s += p[(size_t)n * D_];
    float h = s * (1.0f/(float)N_);
    Hf[v] = h; Hb[v] = f2bf(h);
  }
  __syncthreads();

  const float scale = 0.0625f;               // 1/sqrt(256)
  const size_t TOT = (size_t)B_ * N_;

#pragma unroll 1
  for (int step = 0; step < N_; ++step){
    // ---- stage 1: gate GEMMs. waves 0-7: gi = X@w_ih^T ; waves 8-15: gh = H@w_hh^T
    {
      const u16* Asrc = (wave < 8) ? Xb   : Hb;
      const u16* Wsrc = (wave < 8) ? wihb : whhb;
      float*     Gdst = (wave < 8) ? GI   : GH;
      const int  w8   = wave & 7;
      v16bf a[8];
#pragma unroll
      for (int kc = 0; kc < 8; ++kc)
        a[kc] = load_frag(Asrc + lrow*D_ + kc*32, hlf);
#pragma unroll 1
      for (int q = 0; q < 6; ++q){
        int t = w8*6 + q;                    // 48 N-tiles over 8 waves
        v8f acc = {0.f,0.f,0.f,0.f,0.f,0.f,0.f,0.f};
#pragma unroll
        for (int kc = 0; kc < 8; ++kc)
          acc = wmma_bf16(a[kc], load_frag(Wsrc + (t*16 + lrow)*D_ + kc*32, hlf), acc);
#pragma unroll
        for (int r = 0; r < 8; ++r)
          Gdst[(r + 8*hlf)*768 + t*16 + lrow] = acc[r];
      }
    }
    __syncthreads();

    // ---- stage 2: elementwise GRU update ----
#pragma unroll 1
    for (int j = 0; j < 8; ++j){
      int v = tid + 512*j;
      int b = v >> 8, d = v & 255;
      float ir  = GI[b*768 + d]       + b_ih[d];
      float iz  = GI[b*768 + 256 + d] + b_ih[256 + d];
      float inn = GI[b*768 + 512 + d] + b_ih[512 + d];
      float hr  = GH[b*768 + d]       + b_hh[d];
      float hz  = GH[b*768 + 256 + d] + b_hh[256 + d];
      float hn  = GH[b*768 + 512 + d] + b_hh[512 + d];
      float r  = sigmoidf_(ir + hr);
      float z  = sigmoidf_(iz + hz);
      float nn = tanhf(inn + r*hn);
      float h  = (1.0f - z)*nn + z*Hf[v];
      Hf[v] = h; Hb[v] = f2bf(h);
    }
    __syncthreads();

    // ---- stage 3: query = H @ wq^T + bq  (1 tile per wave) ----
    {
      const int t = wave;
      v16bf a[8];
#pragma unroll
      for (int kc = 0; kc < 8; ++kc)
        a[kc] = load_frag(Hb + lrow*D_ + kc*32, hlf);
      v8f acc = {0.f,0.f,0.f,0.f,0.f,0.f,0.f,0.f};
#pragma unroll
      for (int kc = 0; kc < 8; ++kc)
        acc = wmma_bf16(a[kc], load_frag(wqb + (t*16 + lrow)*D_ + kc*32, hlf), acc);
      float bqv = bq[t*16 + lrow];
#pragma unroll
      for (int r = 0; r < 8; ++r)
        Qb[(r + 8*hlf)*D_ + t*16 + lrow] = f2bf(acc[r] + bqv);
    }
    __syncthreads();

    // ---- stage 4: logits[b][n] = scale * q[b] . keys[b][n]  (1 batch per wave)
    {
      const int b = wave;
      const u16* kb = keysb + ((size_t)(bg0 + b)) * (size_t)N_ * D_;
      v16bf qf[8];                           // broadcast B operand (query chunks)
#pragma unroll
      for (int kc = 0; kc < 8; ++kc)
        qf[kc] = load_frag(Qb + b*D_ + kc*32, hlf);
#pragma unroll 1
      for (int t = 0; t < 16; ++t){
        if (t < 15) __builtin_prefetch(kb + (size_t)(t+1)*16*D_ + lane*128, 0, 1);
        v8f acc = {0.f,0.f,0.f,0.f,0.f,0.f,0.f,0.f};
#pragma unroll
        for (int kc = 0; kc < 8; ++kc)
          acc = wmma_bf16(load_frag(kb + (t*16 + lrow)*D_ + kc*32, hlf), qf[kc], acc);
        if (lrow == 0){                      // all D columns equal -> lanes 0/16 emit
#pragma unroll
          for (int r = 0; r < 8; ++r){
            int n = t*16 + 8*hlf + r;
            LG[b*D_ + n] = MK[b*D_ + n] ? acc[r]*scale : NEG_INF_;
          }
        }
      }
    }
    __syncthreads();

    // ---- stage 5: masked log-softmax / argmax / entropy / select (2 b per wave)
    if (wave < 8){
      const int b  = 2*wave + hlf;
      const int bg = bg0 + b;
      const float* lg = LG + b*D_;
      float best = -3.402823e38f; int bidx = 0;
#pragma unroll 1
      for (int i = 0; i < 16; ++i){
        int n = lrow*16 + i; float v = lg[n];
        if (v > best){ best = v; bidx = n; }     // first occurrence wins
      }
#pragma unroll
      for (int m = 8; m >= 1; m >>= 1){
        float ov = __shfl_xor(best, m, 32);
        int   oi = __shfl_xor(bidx, m, 32);
        if (ov > best || (ov == best && oi < bidx)){ best = ov; bidx = oi; }
      }
      float se = 0.f;
#pragma unroll 1
      for (int i = 0; i < 16; ++i) se += expf(lg[lrow*16 + i] - best);
#pragma unroll
      for (int m = 8; m >= 1; m >>= 1) se += __shfl_xor(se, m, 32);
      float logZ = best + logf(se);
      float ent = 0.f;
#pragma unroll 1
      for (int i = 0; i < 16; ++i){
        int n = lrow*16 + i;
        float lp = lg[n] - logZ;
        if (MK[b*D_ + n]) ent += expf(lp)*lp;
      }
#pragma unroll
      for (int m = 8; m >= 1; m >>= 1) ent += __shfl_xor(ent, m, 32);
      ent = -ent;
      if (lrow == 0){
        out[(size_t)bg*N_ + step]         = (float)bidx;
        out[TOT   + (size_t)bg*N_ + step] = lg[bidx] - logZ;
        out[2*TOT + (size_t)bg*N_ + step] = ent;
        MK[b*D_ + bidx] = 0;
      }
      // gather next prev_embed (bf16) from enc
      const u16* src = encb + ((size_t)bg*N_ + bidx)*D_;
#pragma unroll
      for (int i = 0; i < 16; ++i)
        Xb[b*D_ + lrow*16 + i] = src[lrow*16 + i];
    }
    __syncthreads();
  }
}

// ---------------- host launcher ----------------
extern "C" void kernel_launch(void* const* d_in, const int* in_sizes, int n_in,
                              void* d_out, int out_size, void* d_ws, size_t ws_size,
                              hipStream_t stream) {
  (void)in_sizes; (void)n_in; (void)out_size; (void)ws_size;
  const float* enc   = (const float*)d_in[0];
  const float* initt = (const float*)d_in[1];
  const float* w_ih  = (const float*)d_in[2];
  const float* w_hh  = (const float*)d_in[3];
  const float* b_ih  = (const float*)d_in[4];
  const float* b_hh  = (const float*)d_in[5];
  const float* wq    = (const float*)d_in[6];
  const float* bq    = (const float*)d_in[7];
  const float* wk    = (const float*)d_in[8];
  const float* bk    = (const float*)d_in[9];
  float* out = (float*)d_out;

  // workspace layout (bf16 copies): enc, keys, w_ih, w_hh, wq, wk  (~135 MB)
  u16* ws = (u16*)d_ws;
  const size_t encN = (size_t)B_ * N_ * D_;     // 33,554,432 elements
  u16* encb  = ws;
  u16* keysb = encb  + encN;
  u16* wihb  = keysb + encN;
  u16* whhb  = wihb  + (size_t)3*D_*D_;
  u16* wqb   = whhb  + (size_t)3*D_*D_;
  u16* wkb   = wqb   + (size_t)D_*D_;

  k_to_bf16<<<4096, 256, 0, stream>>>(enc,  encb, (int)encN);
  k_to_bf16<<<192,  256, 0, stream>>>(w_ih, wihb, 3*D_*D_);
  k_to_bf16<<<192,  256, 0, stream>>>(w_hh, whhb, 3*D_*D_);
  k_to_bf16<<<64,   256, 0, stream>>>(wq,   wqb,  D_*D_);
  k_to_bf16<<<64,   256, 0, stream>>>(wk,   wkb,  D_*D_);

  k_keys_gemm<<<(B_*N_)/128, 256, 0, stream>>>(encb, wkb, bk, keysb);

  const size_t smem = (size_t)(3*16*256*sizeof(u16)   // Xb,Hb,Qb
                     + 16*256*sizeof(float)           // Hf
                     + 2*16*768*sizeof(float)         // GI,GH
                     + 16*256*sizeof(float)           // LG
                     + 16*256);                       // MK  => 159,744 B
  k_decode<<<B_/16, 512, smem, stream>>>(enc, initt, b_ih, b_hh, bq,
                                         encb, keysb, wihb, whhb, wqb, out);
}
